// AttentionGeo_49108656063069
// MI455X (gfx1250) — matrix operand order repaired and checked by reading, not compiled, not run on hardware
//
#include <hip/hip_runtime.h>
#include <hip/hip_bf16.h>

typedef __attribute__((ext_vector_type(16))) _Float16 v16h;
typedef __attribute__((ext_vector_type(8)))  float    v8f;

namespace {
constexpr int KNB  = 50;   // neighbors
constexpr int JTOT = 51;   // node2vec rows per batch
constexpr int FDIM = 80;   // context features
constexpr int DDIM = 64;   // embedding dim
constexpr int BB   = 16;   // batch rows per block (one WMMA M-tile)
constexpr int BDIM = 256;  // 8 wave32
constexpr int WAVES = 8;

// dynamic-LDS layout (bytes); all chunks 16B aligned
constexpr int OFF_WFRAG = 0;                        // 4 matrices * 4096 f16 (B-fragment order)
constexpr int OFF_WB    = OFF_WFRAG + 4 * 4096 * 2; // 4 * 64 f32 biases
constexpr int OFF_TGT   = OFF_WB    + 4 * 64 * 4;   // 16 x 64 f32 normalized targets
constexpr int OFF_SIMI  = OFF_TGT   + 16 * 64 * 4;  // 16 x 50 f32
constexpr int OFF_LOG   = OFF_SIMI  + 16 * 50 * 4;  // 16 x 50 f32 logits/weights
constexpr int OFF_STAGE = OFF_LOG   + 16 * 50 * 4;  // per-wave 16x64 f32 (aliased f16 view)
constexpr int LDS_TOTAL = OFF_STAGE + WAVES * 16 * 64 * 4;
}

// A fragment (16x32 f16): lane<16 holds K in {0..7,16..23}, lane>=16 holds {8..15,24..31}
__device__ __forceinline__ v16h loadA(const _Float16* st, int lane, int kt) {
  union { v16h h; uint4 u[2]; } a;
  const char* base = (const char*)(st + (lane & 15) * 64 + kt * 32 + ((lane >> 4) << 3));
  a.u[0] = *(const uint4*)(base);
  a.u[1] = *(const uint4*)(base + 32);
  return a.h;
}

// B fragment: pre-swizzled contiguous 16 halves per lane
__device__ __forceinline__ v16h loadB(const _Float16* wf, int lane, int frag) {
  union { v16h h; uint4 u[2]; } b;
  const uint4* p = (const uint4*)(wf + (frag * 32 + lane) * 16);
  b.u[0] = p[0];
  b.u[1] = p[1];
  return b.h;
}

// one 16x64 @ 64x64 layer: D = relu?(X*W + bias). bias folded into accumulator init.
template <bool RELU>
__device__ __forceinline__ void layer64(const _Float16* xin, const _Float16* wf,
                                        const float* bias, int lane,
                                        _Float16* out16, float* out32) {
  v16h a0 = loadA(xin, lane, 0);
  v16h a1 = loadA(xin, lane, 1);
  __builtin_amdgcn_wave_barrier();   // a-frags in regs before out buffer (aliased) is written
  const int n0  = lane & 15;
  const int mhi = (lane >> 4) << 3;  // C frag: rows M = mhi + v
#pragma unroll
  for (int nt = 0; nt < 4; ++nt) {
    v16h b0 = loadB(wf, lane, nt);        // kt=0 fragments 0..3
    v16h b1 = loadB(wf, lane, 4 + nt);    // kt=1 fragments 4..7
    float bv = bias[nt * 16 + n0];
    v8f c;
#pragma unroll
    for (int v = 0; v < 8; ++v) c[v] = bv;
    c = __builtin_amdgcn_wmma_f32_16x16x32_f16(false, a0, false, b0, (short)0, c, false, false);
    c = __builtin_amdgcn_wmma_f32_16x16x32_f16(false, a1, false, b1, (short)0, c, false, false);
#pragma unroll
    for (int v = 0; v < 8; ++v) {
      int m = mhi + v;
      int col = nt * 16 + n0;
      float x = c[v];
      if (RELU) out16[m * 64 + col] = (_Float16)(x > 0.f ? x : 0.f);
      else      out32[m * 64 + col] = x;
    }
  }
}

// full 2-layer MLP on a 16-row tile; result f32 in st32 (same LDS as st16)
__device__ __forceinline__ void mlp_tile(const float* xg, int rowstride,
                                         const _Float16* wfa, const _Float16* wfb,
                                         const float* b1, const float* b2,
                                         _Float16* st16, float* st32, int lane) {
#pragma unroll
  for (int t = 0; t < 32; ++t) {   // 16x64 elements, coalesced, f32 -> f16
    int e = lane + t * 32;
    st16[e] = (_Float16)xg[(e >> 6) * rowstride + (e & 63)];
  }
  __builtin_amdgcn_wave_barrier();
  layer64<true >(st16, wfa, b1, lane, st16, nullptr);
  __builtin_amdgcn_wave_barrier();
  layer64<false>(st16, wfb, b2, lane, nullptr, st32);
  __builtin_amdgcn_wave_barrier();
}

__global__ __launch_bounds__(BDIM) void attgeo_kernel(
    const float* __restrict__ sd, const float* __restrict__ ctx,
    const float* __restrict__ n2v,
    const float* __restrict__ w11, const float* __restrict__ b11,
    const float* __restrict__ w12, const float* __restrict__ b12,
    const float* __restrict__ w21, const float* __restrict__ b21,
    const float* __restrict__ w22, const float* __restrict__ b22,
    const float* __restrict__ kern, const float* __restrict__ kbias,
    float* __restrict__ out, int B) {
  extern __shared__ char smem[];
  _Float16* wfrag = (_Float16*)(smem + OFF_WFRAG);
  float* wb   = (float*)(smem + OFF_WB);
  float* tgt  = (float*)(smem + OFF_TGT);
  float* simi = (float*)(smem + OFF_SIMI);
  float* lgts = (float*)(smem + OFF_LOG);

  const int tid  = threadIdx.x;
  const int wv   = tid >> 5;
  const int lane = tid & 31;
  const int b0   = blockIdx.x * BB;

  // ---- Phase 0: swizzle the 4 weight matrices into WMMA B-fragment order (f16) ----
  {
    const float* Ws[4] = {w11, w12, w21, w22};
#pragma unroll
    for (int m = 0; m < 4; ++m) {
      const float* W = Ws[m];
      _Float16* dst = wfrag + m * 4096;
      for (int idx = tid; idx < 4096; idx += BDIM) {
        int frag = idx >> 9;           // kt*4+nt
        int ln   = (idx >> 4) & 31;
        int h    = idx & 15;
        int kt = frag >> 2, nt = frag & 3;
        int Kr = kt * 32 + ((ln >> 4) << 3) + (h & 7) + ((h >> 3) << 4);   // B frag: lane<16 -> K {0..7,16..23}, else {8..15,24..31}
        int Nc = nt * 16 + (ln & 15);
        dst[idx] = (_Float16)W[Kr * 64 + Nc];
      }
    }
    const float* Bs[4] = {b11, b12, b21, b22};
    for (int idx = tid; idx < 256; idx += BDIM) wb[idx] = Bs[idx >> 6][idx & 63];
  }
  __syncthreads();

  _Float16* st16 = (_Float16*)(smem + OFF_STAGE + wv * (16 * 64 * 4));
  float*    st32 = (float*)   (smem + OFF_STAGE + wv * (16 * 64 * 4));

  // ---- Phase 1a: MLP1 on row j=0 -> normalized targets (wave 0 only) ----
  if (wv == 0) {
    const float* xg = n2v + (size_t)b0 * (JTOT * DDIM);
    mlp_tile(xg, JTOT * DDIM, wfrag, wfrag + 4096, wb, wb + 64, st16, st32, lane);
    if (lane < 16) {
      const float* row = st32 + lane * 64;
      float ss = 0.f;
      for (int c = 0; c < 64; ++c) ss += row[c] * row[c];
      float sc = rsqrtf(fmaxf(ss, 1e-12f));
      float* trow = tgt + lane * 64;
      for (int c = 0; c < 64; ++c) trow[c] = row[c] * sc;
    }
  }
  __syncthreads();

  // ---- Phase 1b: MLP2 on rows j=1..50; fused norm + dot + gaussian term ----
  for (int j = 1 + wv; j <= KNB; j += WAVES) {
    const float* xg = n2v + ((size_t)b0 * JTOT + j) * DDIM;
    mlp_tile(xg, JTOT * DDIM, wfrag + 2 * 4096, wfrag + 3 * 4096, wb + 128, wb + 192,
             st16, st32, lane);
    if (lane < 16) {
      int k = j - 1;
      const float* row  = st32 + lane * 64;
      const float* trow = tgt + lane * 64;
      float ss = 0.f, dt = 0.f;
      for (int c = 0; c < 64; ++c) { float x = row[c]; ss += x * x; dt += trow[c] * x; }
      float s2 = dt * rsqrtf(fmaxf(ss, 1e-12f)) * (1.0f / 64.0f);  // mean(t * n_hat)
      float d  = sd[(size_t)(b0 + lane) * KNB + k];
      simi[lane * KNB + k] = __expf(-d * d) + 0.1f * s2;
    }
  }
  __syncthreads();

  // ---- Phase 2: logits = simi @ kernel + bias (kernel stays hot in L2) ----
  for (int idx = tid; idx < BB * KNB; idx += BDIM) {
    int i = idx / KNB, m = idx - i * KNB;
    float acc = kbias[m];
    const float* srow = simi + i * KNB;
    for (int k = 0; k < KNB; ++k) acc += srow[k] * kern[k * KNB + m];
    lgts[idx] = acc;
  }
  __syncthreads();
  // softmax per batch row (16 rows)
  for (int i = tid; i < BB; i += BDIM) {
    float* lrow = lgts + i * KNB;
    float mx = lrow[0];
    for (int k = 1; k < KNB; ++k) mx = fmaxf(mx, lrow[k]);
    float s = 0.f;
    for (int k = 0; k < KNB; ++k) { float e = __expf(lrow[k] - mx); lrow[k] = e; s += e; }
    float inv = 1.f / s;
    for (int k = 0; k < KNB; ++k) lrow[k] *= inv;
  }
  __syncthreads();

  // ---- Phase 3: out[b,f] = sum_k w[b,k] * context[b,k,f] (streams 524 MB total) ----
  for (int idx = tid; idx < BB * FDIM; idx += BDIM) {
    int i = idx / FDIM, f = idx - i * FDIM;
    const float* lrow = lgts + i * KNB;
    const float* crow = ctx + ((size_t)(b0 + i) * KNB) * FDIM + f;
    float acc = 0.f;
#pragma unroll 5
    for (int k = 0; k < KNB; ++k) acc += lrow[k] * crow[(size_t)k * FDIM];
    out[(size_t)(b0 + i) * FDIM + f] = acc;
  }
}

extern "C" void kernel_launch(void* const* d_in, const int* in_sizes, int n_in,
                              void* d_out, int out_size, void* d_ws, size_t ws_size,
                              hipStream_t stream) {
  const float* sd    = (const float*)d_in[0];
  const float* ctx   = (const float*)d_in[1];
  const float* n2v   = (const float*)d_in[2];
  const float* w11   = (const float*)d_in[3];
  const float* b11   = (const float*)d_in[4];
  const float* w12   = (const float*)d_in[5];
  const float* b12   = (const float*)d_in[6];
  const float* w21   = (const float*)d_in[7];
  const float* b21   = (const float*)d_in[8];
  const float* w22   = (const float*)d_in[9];
  const float* b22   = (const float*)d_in[10];
  const float* kern  = (const float*)d_in[11];
  const float* kbias = (const float*)d_in[12];
  float* out = (float*)d_out;

  int B = in_sizes[0] / KNB;          // 32768 (multiple of BB)
  dim3 grid((B + BB - 1) / BB), block(BDIM);
  attgeo_kernel<<<grid, block, (size_t)LDS_TOTAL, stream>>>(
      sd, ctx, n2v, w11, b11, w12, b12, w21, b21, w22, b22, kern, kbias, out, B);
}
